// MaskedMultiHeadSelfAttention_77309412031
// MI455X (gfx1250) — compile-verified, matrix operand order checked by reading, not run on hardware
//
#include <hip/hip_runtime.h>
#include <hip/hip_bf16.h>

typedef __attribute__((ext_vector_type(16))) _Float16 v16h;
typedef __attribute__((ext_vector_type(8)))  _Float16 v8h;
typedef __attribute__((ext_vector_type(8)))  float    v8f;
typedef __attribute__((ext_vector_type(4)))  float    v4f;

#define NUM_HEADS 16
#define DH        128
#define DMODEL    2048
#define SEQ       2048
#define BATCH     2
#define LDSPITCH  40   // 32 halfs + 8 pad: 16B-aligned, spreads LDS banks

#if defined(__gfx1250__) && __has_builtin(__builtin_amdgcn_global_load_async_to_lds_b128)
#define HAVE_ASYNC_LDS 1
#else
#define HAVE_ASYNC_LDS 0
#endif

// parameter type of the async-LDS builtin: 16-byte int vector
typedef int v4i_t __attribute__((vector_size(4 * sizeof(int))));
typedef __attribute__((address_space(1))) v4i_t gv4i;   // global
typedef __attribute__((address_space(3))) v4i_t lv4i;   // LDS

// Stage one 16B chunk global -> LDS. Async path bypasses VGPRs (ASYNCcnt).
__device__ __forceinline__ void stage16B(const _Float16* gsrc, _Float16* ldst) {
#if HAVE_ASYNC_LDS
  __builtin_amdgcn_global_load_async_to_lds_b128((gv4i*)gsrc, (lv4i*)ldst, 0, 0);
#else
  *(v8h*)ldst = *(const v8h*)gsrc;
#endif
}

__device__ __forceinline__ void wait_async_all() {
#if HAVE_ASYNC_LDS
#if __has_builtin(__builtin_amdgcn_s_wait_asynccnt)
  __builtin_amdgcn_s_wait_asynccnt(0);
#else
  asm volatile("s_wait_asynccnt 0x0" ::: "memory");
#endif
#endif
}

// Split workgroup barrier WITHOUT the full memory fence of __syncthreads():
// keeps in-flight global loads outstanding across the barrier (only asynccnt
// must be drained first, done by wait_async_all()).
__device__ __forceinline__ void tile_barrier() {
#if HAVE_ASYNC_LDS
  asm volatile("s_barrier_signal -1\n\ts_barrier_wait -1" ::: "memory");
#else
  __syncthreads();
#endif
}

// Load a 16x32 f16 fragment (A-layout; also B when the matrix "column" is
// contiguous in memory). ISA layout: lanes 0-15 row=lane, K=0..7 & 16..23;
// lanes 16-31 row=lane-16, K=8..15 & 24..31.
__device__ __forceinline__ v16h load_frag16(const _Float16* __restrict__ base,
                                            int stride, int row, int k0, int lane) {
  const int ko = (lane >> 4) << 3;  // 0 or 8
  const _Float16* p = base + (size_t)row * stride + k0;
  v8h lo = *(const v8h*)(p + ko);
  v8h hi = *(const v8h*)(p + 16 + ko);
  v16h r;
#pragma unroll
  for (int i = 0; i < 8; ++i) { r[i] = lo[i]; r[8 + i] = hi[i]; }
  return r;
}

// Read one B-fragment from an LDS-staged 64x32 tile.
__device__ __forceinline__ v16h lds_frag16(const _Float16* tile, int row, int ko) {
  const _Float16* pr = tile + row * LDSPITCH;
  v8h lo = *(const v8h*)(pr + ko);
  v8h hi = *(const v8h*)(pr + 16 + ko);
  v16h r;
#pragma unroll
  for (int i = 0; i < 8; ++i) { r[i] = lo[i]; r[8 + i] = hi[i]; }
  return r;
}

__global__ void cast_f32_to_f16(const float* __restrict__ src,
                                _Float16* __restrict__ dst, size_t n8) {
  size_t i = (size_t)blockIdx.x * blockDim.x + threadIdx.x;
  if (i >= n8) return;
  v4f a = ((const v4f*)src)[2 * i];
  v4f b = ((const v4f*)src)[2 * i + 1];
  v8h o;
#pragma unroll
  for (int j = 0; j < 4; ++j) { o[j] = (_Float16)a[j]; o[4 + j] = (_Float16)b[j]; }
  ((v8h*)dst)[i] = o;
}

// y[m,n] = sum_k A[m,k] * Wt[n,k] + bias[n]
// 256 threads = 8 waves on a 128(M) x 64(N) tile. 64x32 weight tile staged in
// LDS via async global->LDS copies; k-loop manually unrolled by 2 so the
// ping-pong buffers and the A-fragment double-buffer are static (no copies).
// mode 0: f16 head-split [B,H,S,dh] (*scale);  mode 1: fp32 flat [M,N];
// mode 2: f16 transposed head-split [B,H,dh,S] (for V).
__global__ void __launch_bounds__(256)
wmma_gemm(const _Float16* __restrict__ A,    // [M,K]
          const _Float16* __restrict__ Wt,   // [N,K]
          const float*    __restrict__ bias, // [N]
          void* __restrict__ dst,
          int M, int N, int K, int mode, float scale) {
  __shared__ __align__(16) _Float16 bstage[2][64 * LDSPITCH];
  const int tid  = threadIdx.x;
  const int lane = tid & 31;
  const int wv   = tid >> 5;              // wave 0..7
  const int nt64 = N >> 6;
  const int bm   = blockIdx.x / nt64;     // 128-row block
  const int n64  = blockIdx.x % nt64;
  const int nbase = n64 * 64;
  const int col  = lane & 15;
  const int ko   = (lane >> 4) << 3;
  const int mt   = bm * 8 + wv;           // this wave's 16-row tile
  const int arow = mt * 16 + col;

  // cooperative weight staging: each thread owns one 16B chunk of the tile
  const int srow = tid >> 2;              // 0..63
  const int schk = (tid & 3) << 3;        // half offset 0/8/16/24
  const _Float16* wsrc = Wt + (size_t)(nbase + srow) * K + schk;
  _Float16* ld0 = &bstage[0][srow * LDSPITCH + schk];
  _Float16* ld1 = &bstage[1][srow * LDSPITCH + schk];

  v8f acc[4];
#pragma unroll
  for (int t = 0; t < 4; ++t)
#pragma unroll
    for (int r = 0; r < 8; ++r) acc[t][r] = 0.f;

  // prologue: stage k-step 0 + preload first A fragment
  stage16B(wsrc, ld0);
  v16h a0 = load_frag16(A, K, arow, 0, lane);
  wait_async_all();
  __syncthreads();

  for (int kk = 0; kk < K; kk += 64) {   // K % 64 == 0
    // ---- half 1: compute from bstage[0]/a0, stage bstage[1], load a1 ----
    stage16B(wsrc + kk + 32, ld1);       // kk+32 < K always (K mult of 64)
    __builtin_prefetch(A + (size_t)arow * K + kk + 512, 0, 1);
    v16h a1 = load_frag16(A, K, arow, kk + 32, lane);
#pragma unroll
    for (int t = 0; t < 4; ++t)
      acc[t] = __builtin_amdgcn_wmma_f32_16x16x32_f16(
          false, a0, false, lds_frag16(bstage[0], t * 16 + col, ko),
          (short)0, acc[t], false, false);
    wait_async_all();
    tile_barrier();

    // ---- half 2: compute from bstage[1]/a1, stage bstage[0], load a0 ----
    const bool more = (kk + 64 < K);     // uniform
    if (more) stage16B(wsrc + kk + 64, ld0);
    const int ka = more ? kk + 64 : 0;   // clamped: stays in-bounds
    a0 = load_frag16(A, K, arow, ka, lane);
#pragma unroll
    for (int t = 0; t < 4; ++t)
      acc[t] = __builtin_amdgcn_wmma_f32_16x16x32_f16(
          false, a1, false, lds_frag16(bstage[1], t * 16 + col, ko),
          (short)0, acc[t], false, false);
    if (more) {
      wait_async_all();
      tile_barrier();
    }
  }

  const int rbase = (lane >> 4) << 3;
#pragma unroll
  for (int t = 0; t < 4; ++t) {
    const int n  = nbase + t * 16 + col;
    const float bv = bias[n];
#pragma unroll
    for (int r = 0; r < 8; ++r) {
      const int   m   = mt * 16 + rbase + r;
      const float val = acc[t][r] + bv;
      const int b = m / SEQ;
      const int s = m - b * SEQ;
      const int h = n >> 7;
      const int d = n & (DH - 1);
      if (mode == 0) {
        ((_Float16*)dst)[(((size_t)(b * NUM_HEADS + h) * SEQ + s) * DH) + d] =
            (_Float16)(val * scale);
      } else if (mode == 2) {
        ((_Float16*)dst)[(((size_t)(b * NUM_HEADS + h) * DH + d) * SEQ) + s] =
            (_Float16)val;
      } else {
        ((float*)dst)[(size_t)m * N + n] = val;
      }
    }
  }
}

// One wave per (b*H+h, 16-query tile). Streaming causal flash attention.
// V is consumed transposed ([B,H,dh,S]) so P*V B-fragments are two 16B loads.
__global__ void __launch_bounds__(32)
flash_attn(const _Float16* __restrict__ Qh,   // [B,H,S,dh], pre-scaled 1/sqrt(dh)
           const _Float16* __restrict__ Kh,   // [B,H,S,dh]
           const _Float16* __restrict__ Vt,   // [B,H,dh,S]
           _Float16* __restrict__ Out) {      // [B,S,D]
  __shared__ __align__(16) _Float16 lds[16 * 32];
  const int lane = threadIdx.x & 31;
  const int col  = lane & 15;
  const int grp  = lane >> 4;
  const int ko   = grp << 3;
  const int qtiles = SEQ >> 4;
  const int bh = blockIdx.x / qtiles;
  const int qt = blockIdx.x % qtiles;
  const size_t base  = (size_t)bh * SEQ * DH;
  const int    qbase = qt * 16;

  v16h qa[4];
#pragma unroll
  for (int kt = 0; kt < 4; ++kt)
    qa[kt] = load_frag16(Qh + base, DH, qbase + col, kt * 32, lane);

  v8f acc[8];
  float mrow[8], lrow[8];
#pragma unroll
  for (int t = 0; t < 8; ++t)
#pragma unroll
    for (int r = 0; r < 8; ++r) acc[t][r] = 0.f;
#pragma unroll
  for (int r = 0; r < 8; ++r) { mrow[r] = -1e30f; lrow[r] = 0.f; }

  const int kend = qbase + 16;  // exclusive causal key bound
  for (int k0 = 0; k0 < kend; k0 += 32) {
    // ---- all 8 K-fragments up front (one wait), then 8 score WMMAs ----
    v16h kb[8];
#pragma unroll
    for (int kt = 0; kt < 4; ++kt) {
      kb[kt]     = load_frag16(Kh + base, DH, k0 + col,      kt * 32, lane);
      kb[4 + kt] = load_frag16(Kh + base, DH, k0 + 16 + col, kt * 32, lane);
    }
    v8f s0, s1;
#pragma unroll
    for (int r = 0; r < 8; ++r) { s0[r] = 0.f; s1[r] = 0.f; }
#pragma unroll
    for (int kt = 0; kt < 4; ++kt) {
      s0 = __builtin_amdgcn_wmma_f32_16x16x32_f16(
          false, qa[kt], false, kb[kt], (short)0, s0, false, false);
      s1 = __builtin_amdgcn_wmma_f32_16x16x32_f16(
          false, qa[kt], false, kb[4 + kt], (short)0, s1, false, false);
    }
    // ---- causal mask ----
#pragma unroll
    for (int r = 0; r < 8; ++r) {
      const int q = qbase + (grp << 3) + r;
      if (k0 + col      > q) s0[r] = -1e30f;
      if (k0 + 16 + col > q) s1[r] = -1e30f;
    }
    // ---- online softmax row stats (16-lane group butterflies) ----
    float rmax[8], rsum[8], sc[8];
#pragma unroll
    for (int r = 0; r < 8; ++r) rmax[r] = fmaxf(s0[r], s1[r]);
#pragma unroll
    for (int off = 1; off < 16; off <<= 1)
#pragma unroll
      for (int r = 0; r < 8; ++r)
        rmax[r] = fmaxf(rmax[r], __shfl_xor(rmax[r], off, 32));
#pragma unroll
    for (int r = 0; r < 8; ++r) {
      const float mnew = fmaxf(mrow[r], rmax[r]);
      sc[r]  = __expf(mrow[r] - mnew);
      mrow[r] = mnew;
      s0[r] = __expf(s0[r] - mnew);
      s1[r] = __expf(s1[r] - mnew);
      rsum[r] = s0[r] + s1[r];
    }
#pragma unroll
    for (int off = 1; off < 16; off <<= 1)
#pragma unroll
      for (int r = 0; r < 8; ++r) rsum[r] += __shfl_xor(rsum[r], off, 32);
#pragma unroll
    for (int r = 0; r < 8; ++r) lrow[r] = lrow[r] * sc[r] + rsum[r];
#pragma unroll
    for (int t = 0; t < 8; ++t)
#pragma unroll
      for (int r = 0; r < 8; ++r) acc[t][r] *= sc[r];

    // ---- transpose P (D-layout -> A-layout) through LDS ----
    __syncthreads();
#pragma unroll
    for (int r = 0; r < 8; ++r) {
      const int m = (grp << 3) + r;
      lds[m * 32 + col]      = (_Float16)s0[r];
      lds[m * 32 + 16 + col] = (_Float16)s1[r];
    }
    __syncthreads();
    v16h pa;
    {
      const _Float16* pr = &lds[col * 32];
      v8h lo = *(const v8h*)(pr + ko);
      v8h hi = *(const v8h*)(pr + 16 + ko);
#pragma unroll
      for (int i = 0; i < 8; ++i) { pa[i] = lo[i]; pa[8 + i] = hi[i]; }
    }
    // ---- O += P x V: 2 groups of (4 loads, then 4 WMMAs) ----
#pragma unroll
    for (int g = 0; g < 2; ++g) {
      v16h vb[4];
#pragma unroll
      for (int t = 0; t < 4; ++t)
        vb[t] = load_frag16(Vt + base, SEQ, (g * 4 + t) * 16 + col, k0, lane);
#pragma unroll
      for (int t = 0; t < 4; ++t)
        acc[g * 4 + t] = __builtin_amdgcn_wmma_f32_16x16x32_f16(
            false, pa, false, vb[t], (short)0, acc[g * 4 + t], false, false);
    }
  }

  // ---- normalize and scatter back to [B,S,D] ----
  const int b = bh / NUM_HEADS;
  const int h = bh - b * NUM_HEADS;
#pragma unroll
  for (int t = 0; t < 8; ++t)
#pragma unroll
    for (int r = 0; r < 8; ++r) {
      const int   s = qbase + (grp << 3) + r;
      const float o = acc[t][r] / lrow[r];
      Out[(size_t)(b * SEQ + s) * DMODEL + h * DH + t * 16 + col] = (_Float16)o;
    }
}

extern "C" void kernel_launch(void* const* d_in, const int* in_sizes, int n_in,
                              void* d_out, int out_size, void* d_ws, size_t ws_size,
                              hipStream_t stream) {
  const float* x  = (const float*)d_in[0];
  const float* Wq = (const float*)d_in[1];
  const float* bq = (const float*)d_in[2];
  const float* Wk = (const float*)d_in[3];
  const float* bk = (const float*)d_in[4];
  const float* Wv = (const float*)d_in[5];
  const float* bv = (const float*)d_in[6];
  const float* Wo = (const float*)d_in[7];
  const float* bo = (const float*)d_in[8];

  const size_t NX = (size_t)BATCH * SEQ * DMODEL;
  const size_t NW = (size_t)DMODEL * DMODEL;

  _Float16* p   = (_Float16*)d_ws;
  _Float16* xh  = p; p += NX;
  _Float16* wqh = p; p += NW;
  _Float16* wkh = p; p += NW;
  _Float16* wvh = p; p += NW;
  _Float16* woh = p; p += NW;
  _Float16* Qh  = p; p += NX;   // [B,H,S,dh]
  _Float16* Kh  = p; p += NX;   // [B,H,S,dh]
  _Float16* Vt  = p; p += NX;   // [B,H,dh,S]
  _Float16* Ah  = p; p += NX;   // attention out, [B,S,D]

  const int CT = 256;
  cast_f32_to_f16<<<dim3((unsigned)((NX / 8 + CT - 1) / CT)), dim3(CT), 0, stream>>>(x,  xh,  NX / 8);
  cast_f32_to_f16<<<dim3((unsigned)((NW / 8 + CT - 1) / CT)), dim3(CT), 0, stream>>>(Wq, wqh, NW / 8);
  cast_f32_to_f16<<<dim3((unsigned)((NW / 8 + CT - 1) / CT)), dim3(CT), 0, stream>>>(Wk, wkh, NW / 8);
  cast_f32_to_f16<<<dim3((unsigned)((NW / 8 + CT - 1) / CT)), dim3(CT), 0, stream>>>(Wv, wvh, NW / 8);
  cast_f32_to_f16<<<dim3((unsigned)((NW / 8 + CT - 1) / CT)), dim3(CT), 0, stream>>>(Wo, woh, NW / 8);

  const int M = BATCH * SEQ;   // 4096
  const int N = DMODEL;        // 2048
  const int K = DMODEL;        // 2048
  const dim3 gblk(256);
  const dim3 ggrd((unsigned)((M / 128) * (N / 64)));   // 1024 blocks

  const float qscale = 0.08838834764831845f;  // 1/sqrt(128), folded into Q

  wmma_gemm<<<ggrd, gblk, 0, stream>>>(xh, wqh, bq, Qh, M, N, K, 0, qscale);
  wmma_gemm<<<ggrd, gblk, 0, stream>>>(xh, wkh, bk, Kh, M, N, K, 0, 1.0f);
  wmma_gemm<<<ggrd, gblk, 0, stream>>>(xh, wvh, bv, Vt, M, N, K, 2, 1.0f);

  flash_attn<<<dim3(BATCH * NUM_HEADS * (SEQ / 16)), dim3(32), 0, stream>>>(Qh, Kh, Vt, Ah);

  wmma_gemm<<<ggrd, gblk, 0, stream>>>(Ah, woh, bo, d_out, M, N, K, 1, 1.0f);
}
